// Custom_SelfAttentionBlock_76854144794843
// MI455X (gfx1250) — matrix operand order repaired and checked by reading, not compiled, not run on hardware
//
#include <hip/hip_runtime.h>
#include <hip/hip_bf16.h>
#include <math.h>

// ---------------------------------------------------------------------------
// CDNA5 (gfx1250) wave32 WMMA self-attention block.
// bf16 WMMA everywhere; GEMM tiles staged by the Tensor Data Mover
// (tensor_load_to_lds, TENSORcnt), attention tiles by async global->LDS DMA
// (ASYNCcnt); flash-attention online softmax in registers.
// ---------------------------------------------------------------------------

typedef __attribute__((ext_vector_type(8)))  __bf16 v8bf;
typedef __attribute__((ext_vector_type(16))) __bf16 v16bf;
typedef __attribute__((ext_vector_type(8)))  float  v8f;

typedef __attribute__((ext_vector_type(4))) unsigned tdm_g0_t;  // D# group 0
typedef __attribute__((ext_vector_type(8))) int      tdm_g1_t;  // D# group 1
typedef __attribute__((ext_vector_type(4))) int      tdm_g2_t;  // groups 2/3
typedef __attribute__((ext_vector_type(8))) int      tdm_g4_t;  // 6-arg extra

__device__ __forceinline__ __bf16 to_bf16(float f) {
  unsigned u = __builtin_bit_cast(unsigned, f);
  u += 0x7FFFu + ((u >> 16) & 1u);           // round-to-nearest-even
  unsigned short s = (unsigned short)(u >> 16);
  return __builtin_bit_cast(__bf16, s);
}

// ---- TDM: one-instruction 2-D tile DMA, global -> LDS (ISA 08, section 8) --
// tile_d0 elements per row (bf16), tile_d1 rows; row stride = stride_d0 elems.
// Optional LDS padding: pad_amount_dw DWORDs inserted every (1<<(code+1)) DW.
__device__ __forceinline__ void tdm_load_2d(unsigned lds_off, const void* gptr,
                                            unsigned tile_d0, unsigned tile_d1,
                                            unsigned stride_d0,
                                            unsigned pad_interval_code,
                                            unsigned pad_amount_dw) {
  unsigned long long ga = (unsigned long long)(size_t)gptr;
  tdm_g0_t g0;
  g0[0] = 1u;                                  // count=1 (valid user D#)
  g0[1] = lds_off;                             // lds_addr (bytes)
  g0[2] = (unsigned)(ga & 0xFFFFFFFFu);        // global_addr[31:0]
  g0[3] = (unsigned)(ga >> 32) | (2u << 30);   // global_addr[56:32] | type=2
  unsigned w0 = (1u << 16);                    // data_size = 1 -> 2 bytes
  if (pad_amount_dw)
    w0 |= (1u << 20)                           // pad_enable
        | (pad_interval_code << 22)            // DWORDs between pads
        | ((pad_amount_dw - 1u) << 25);        // pad amount (DWORDs, minus 1)
  tdm_g1_t g1;
  g1[0] = (int)w0;                                          // mask=0, no bar
  g1[1] = (int)((tile_d0 & 0xFFFFu) << 16);                 // tensor_dim0 lo
  g1[2] = (int)((tile_d0 >> 16) | ((tile_d1 & 0xFFFFu) << 16)); // td0 hi|td1 lo
  g1[3] = (int)((tile_d1 >> 16) | (tile_d0 << 16));         // td1 hi | tile_dim0
  g1[4] = (int)(tile_d1 & 0xFFFFu);                         // tile_dim1; dim2=0
  g1[5] = (int)stride_d0;                                   // dim0 stride lo
  g1[6] = 0;                                                // stride hi / dim1 stride
  g1[7] = 0;
  tdm_g2_t z  = {0, 0, 0, 0};                  // 2-D tensor: groups 2/3 unused
  tdm_g4_t z8 = {0, 0, 0, 0, 0, 0, 0, 0};
  __builtin_amdgcn_tensor_load_to_lds(g0, g1, z, z, z8, 0);
}

// ---- Async DMA: 16B global -> LDS, ASYNCcnt-tracked (no VGPR transit) ------
__device__ __forceinline__ void async_copy16B(void* lds_ptr, const void* gptr) {
  unsigned lds_off = (unsigned)(size_t)lds_ptr;
  unsigned long long gaddr = (unsigned long long)(size_t)gptr;
  asm volatile("global_load_async_to_lds_b128 %0, %1, off"
               :: "v"(lds_off), "v"(gaddr) : "memory");
}
__device__ __forceinline__ void wait_async0() {
  asm volatile("s_wait_asynccnt 0x0" ::: "memory");
}

// Concatenate two aligned 8x-bf16 LDS reads into one WMMA fragment register set.
__device__ __forceinline__ v16bf frag_cat(const __bf16* p0, const __bf16* p1) {
  v8bf a = *reinterpret_cast<const v8bf*>(p0);
  v8bf b = *reinterpret_cast<const v8bf*>(p1);
  return __builtin_shufflevector(a, b, 0,1,2,3,4,5,6,7,8,9,10,11,12,13,14,15);
}

__device__ __forceinline__ float gelu_f(float x) {
  const float c = 0.7978845608028654f;            // sqrt(2/pi)
  float y = c * (x + 0.044715f * x * x * x);
  float t = __expf(-2.0f * fabsf(y));             // stable tanh via exp
  float th = (1.0f - t) / (1.0f + t);
  th = copysignf(th, y);
  return 0.5f * x * (1.0f + th);
}

// ---------------------------------------------------------------------------
// f32 -> bf16 converters (plain and transposing, for WMMA-friendly B layout)
// ---------------------------------------------------------------------------
__global__ void cvt_bf16_kernel(const float* __restrict__ in,
                                __bf16* __restrict__ out, int n) {
  int i = blockIdx.x * 256 + threadIdx.x;
  if (i < n) out[i] = to_bf16(in[i]);
}

// in: f32 [R][C] row-major, out: bf16 [C][R]   (B^T for GEMM)
__global__ void cvt_bf16_t_kernel(const float* __restrict__ in,
                                  __bf16* __restrict__ out, int R, int C) {
  int i = blockIdx.x * 256 + threadIdx.x;
  if (i < R * C) {
    int r = i / C, c = i % C;
    out[(size_t)c * R + r] = to_bf16(in[i]);
  }
}

// ---------------------------------------------------------------------------
// Tiled bf16 GEMM:  C[M][N] = A[M][K] @ B  with B supplied transposed [N][K].
// Block tile 128x128, k-step 32. 256 threads = 8 waves as 4(m) x 2(n);
// each wave computes 32x64 = 2x4 WMMA accumulators.
// Double-buffered TDM staging: wave 0 issues one tensor_load_to_lds per tile
// (pad_enable reproduces the LDA=40 padded stride); the next tile's DMA flies
// underneath the current tile's 8 WMMAs; one barrier per k-step.
// ---------------------------------------------------------------------------
#define BM 128
#define BN 128
#define BK 32
#define LDA 40   // BK + 8 bf16 pad: 80B row stride (64B data + 16B TDM pad)

__global__ __launch_bounds__(256)
void gemm_bf16_kernel(const __bf16* __restrict__ A,
                      const __bf16* __restrict__ Bt,
                      const float*  __restrict__ bias,
                      __bf16* __restrict__ outB,
                      float*  __restrict__ outF,
                      int M, int N, int K, int fuse_gelu) {
  __shared__ alignas(16) __bf16 As[2][BM][LDA];
  __shared__ alignas(16) __bf16 Bs[2][BN][LDA];

  const int tid  = threadIdx.x;
  const int lane = tid & 31;
  const int wid  = tid >> 5;
  const int lo   = lane & 15, hi = lane >> 4;
  const int wm   = wid >> 1;          // 0..3
  const int wn   = wid & 1;           // 0..1
  const int m0   = blockIdx.y * BM;
  const int n0   = blockIdx.x * BN;

  v8f acc[2][4];
  const v8f vzero = {0.f,0.f,0.f,0.f,0.f,0.f,0.f,0.f};
#pragma unroll
  for (int am = 0; am < 2; ++am)
#pragma unroll
    for (int bn = 0; bn < 4; ++bn) acc[am][bn] = vzero;

  const __bf16* aTile = A  + (size_t)m0 * K;   // 128 x K panel
  const __bf16* bTile = Bt + (size_t)n0 * K;

  // prologue: TDM tile 0 into buffer 0 (row = 16 DW -> interval code 3; 4 DW pad)
  if (wid == 0) {
    tdm_load_2d((unsigned)(size_t)&As[0][0][0], aTile, BK, BM, (unsigned)K, 3u, 4u);
    tdm_load_2d((unsigned)(size_t)&Bs[0][0][0], bTile, BK, BN, (unsigned)K, 3u, 4u);
  }

  for (int k0 = 0; k0 < K; k0 += BK) {
    const int cur = (k0 / BK) & 1;
    if (wid == 0) __builtin_amdgcn_s_wait_tensorcnt((short)0);  // tile landed
    __syncthreads();      // broadcast completion; prior-tile reads all done
    if (wid == 0 && k0 + BK < K) {   // kick TDM for next tile, other buffer
      tdm_load_2d((unsigned)(size_t)&As[cur ^ 1][0][0], aTile + k0 + BK,
                  BK, BM, (unsigned)K, 3u, 4u);
      tdm_load_2d((unsigned)(size_t)&Bs[cur ^ 1][0][0], bTile + k0 + BK,
                  BK, BN, (unsigned)K, 3u, 4u);
    }

    v16bf af[2], bfq[4];
#pragma unroll
    for (int am = 0; am < 2; ++am) {
      const __bf16* r = &As[cur][wm * 32 + am * 16 + lo][0];
      af[am] = frag_cat(r + hi * 8, r + 16 + hi * 8);
    }
#pragma unroll
    for (int bn = 0; bn < 4; ++bn) {
      const __bf16* r = &Bs[cur][wn * 64 + bn * 16 + lo][0];
      bfq[bn] = frag_cat(r + hi * 8, r + 16 + hi * 8);
    }
#pragma unroll
    for (int am = 0; am < 2; ++am)
#pragma unroll
      for (int bn = 0; bn < 4; ++bn)
        acc[am][bn] = __builtin_amdgcn_wmma_f32_16x16x32_bf16(
            false, af[am], false, bfq[bn], (short)0, acc[am][bn], false, false);
  }

  // Epilogue. C layout: lane(0-15): N=lo, M=r ; lane(16-31): N=lo, M=r+8.
#pragma unroll
  for (int am = 0; am < 2; ++am)
#pragma unroll
    for (int bn = 0; bn < 4; ++bn) {
      const int col = n0 + wn * 64 + bn * 16 + lo;
      const float bv = bias[col];
#pragma unroll
      for (int r = 0; r < 8; ++r) {
        const int row = m0 + wm * 32 + am * 16 + hi * 8 + r;
        float v = acc[am][bn][r] + bv;
        if (fuse_gelu) v = gelu_f(v);
        if (outB) outB[(size_t)row * N + col] = to_bf16(v);
        if (outF) outF[(size_t)row * N + col] = v;
      }
    }
}

// ---------------------------------------------------------------------------
// Flash attention, causal (triu diagonal=1), 8 heads, Dh=64.
// One block = 128 threads (4 waves) handles a 64-row q-tile of one (n,h).
// Each wave owns 16 q-rows; K/V streamed in 32-row tiles. Q/K tiles staged
// by async DMA; V transits VGPRs (needs transposition into [dh][sk]).
// ---------------------------------------------------------------------------
#define AQ 64
#define AK 32
#define DH 64
#define LQS 72   // DH+8  (144B stride)
#define LVS 40   // AK+8  (80B stride)

__global__ __launch_bounds__(128)
void attn_kernel(const __bf16* __restrict__ Q,
                 const __bf16* __restrict__ Kx,
                 const __bf16* __restrict__ V,
                 float* __restrict__ O, int S, int D) {
  __shared__ alignas(16) __bf16 Qs[AQ][LQS];
  __shared__ alignas(16) __bf16 Ks[AK][LQS];
  __shared__ alignas(16) __bf16 Vt[DH][LVS];       // V transposed: [dh][sk]
  __shared__ alignas(16) __bf16 Ps[4][16][LVS];    // per-wave P staging

  const int tid  = threadIdx.x;
  const int lane = tid & 31;
  const int wid  = tid >> 5;
  const int lo   = lane & 15, hi = lane >> 4;
  const int qt = blockIdx.x, h = blockIdx.y, n = blockIdx.z;
  const int qb = qt * AQ;
  const size_t base = ((size_t)n * S) * D + (size_t)h * DH;

  // ---- async DMA of the Q tile (64x64); consumed after first wait+barrier
  {
    int r = tid >> 1, c = (tid & 1) * 32;
    const __bf16* src = Q + base + (size_t)(qb + r) * D + c;
#pragma unroll
    for (int j = 0; j < 4; ++j)
      async_copy16B(&Qs[r][c + 8 * j], src + 8 * j);
  }

  float m_i[8], l_i[8];
#pragma unroll
  for (int r = 0; r < 8; ++r) { m_i[r] = -INFINITY; l_i[r] = 0.0f; }
  v8f o[4];
  const v8f vzero = {0.f,0.f,0.f,0.f,0.f,0.f,0.f,0.f};
#pragma unroll
  for (int t = 0; t < 4; ++t) o[t] = vzero;

  const int ktmax = 2 * qt + 1;               // causal tile bound
  const int mrowb = qb + wid * 16 + hi * 8;   // + r gives this lane's rows

  for (int kt = 0; kt <= ktmax; ++kt) {
    __syncthreads();    // prior iteration's reads of Ks/Vt are complete
    {   // K tile (32x64): async DMA ; V tile (32x64): VGPR transit + transpose
      int r = tid >> 2, c = (tid & 3) * 16;
      const __bf16* ks = Kx + base + (size_t)(kt * AK + r) * D + c;
      async_copy16B(&Ks[r][c],     ks);
      async_copy16B(&Ks[r][c + 8], ks + 8);
      const __bf16* vs = V + base + (size_t)(kt * AK + r) * D + c;
      v8bf v0 = *reinterpret_cast<const v8bf*>(vs);
      v8bf v1 = *reinterpret_cast<const v8bf*>(vs + 8);
#pragma unroll
      for (int j = 0; j < 8; ++j) {
        Vt[c + j][r]     = v0[j];
        Vt[c + 8 + j][r] = v1[j];
      }
    }
    wait_async0();      // own async K (and, first time, Q) DMAs landed
    __syncthreads();    // all waves' tiles visible

    // ---- S = Q @ K^T  (per wave: 16 x 32) ----
    v8f s0 = vzero, s1 = vzero;
#pragma unroll
    for (int kk = 0; kk < 2; ++kk) {
      const __bf16* qr = &Qs[wid * 16 + lo][kk * 32];
      v16bf aq = frag_cat(qr + hi * 8, qr + 16 + hi * 8);
      const __bf16* k0 = &Ks[lo][kk * 32];
      const __bf16* k1 = &Ks[16 + lo][kk * 32];
      v16bf b0 = frag_cat(k0 + hi * 8, k0 + 16 + hi * 8);
      v16bf b1 = frag_cat(k1 + hi * 8, k1 + 16 + hi * 8);
      s0 = __builtin_amdgcn_wmma_f32_16x16x32_bf16(false, aq, false, b0,
                                                   (short)0, s0, false, false);
      s1 = __builtin_amdgcn_wmma_f32_16x16x32_bf16(false, aq, false, b1,
                                                   (short)0, s1, false, false);
    }

    // ---- scale + causal mask ----
    const bool needmask = (kt >= 2 * qt);     // only diagonal-overlap tiles
    const int nc0 = kt * AK + lo, nc1 = kt * AK + 16 + lo;
#pragma unroll
    for (int r = 0; r < 8; ++r) {
      float a0 = s0[r] * 0.125f, a1 = s1[r] * 0.125f;   // 1/sqrt(64)
      if (needmask) {
        if (nc0 > mrowb + r) a0 = -INFINITY;
        if (nc1 > mrowb + r) a1 = -INFINITY;
      }
      s0[r] = a0; s1[r] = a1;
    }

    // ---- online softmax (half-wave reductions match C layout) ----
#pragma unroll
    for (int r = 0; r < 8; ++r) {
      float v = fmaxf(s0[r], s1[r]);
      v = fmaxf(v, __shfl_xor(v, 1, 16));
      v = fmaxf(v, __shfl_xor(v, 2, 16));
      v = fmaxf(v, __shfl_xor(v, 4, 16));
      v = fmaxf(v, __shfl_xor(v, 8, 16));
      const float mnew  = fmaxf(m_i[r], v);
      const float alpha = __expf(m_i[r] - mnew);
      m_i[r] = mnew;
      l_i[r] *= alpha;
#pragma unroll
      for (int t = 0; t < 4; ++t) o[t][r] *= alpha;
      const float p0 = __expf(s0[r] - mnew);
      const float p1 = __expf(s1[r] - mnew);
      float rs = p0 + p1;
      rs += __shfl_xor(rs, 1, 16);
      rs += __shfl_xor(rs, 2, 16);
      rs += __shfl_xor(rs, 4, 16);
      rs += __shfl_xor(rs, 8, 16);
      l_i[r] += rs;
      Ps[wid][hi * 8 + r][lo]      = to_bf16(p0);   // stage P (C -> A layout)
      Ps[wid][hi * 8 + r][16 + lo] = to_bf16(p1);
    }

    // ---- O += P @ V  (A = P 16x32, B = Vt) ----
    const __bf16* pr = &Ps[wid][lo][0];
    v16bf ap = frag_cat(pr + hi * 8, pr + 16 + hi * 8);
#pragma unroll
    for (int t = 0; t < 4; ++t) {
      const __bf16* vr = &Vt[t * 16 + lo][0];
      v16bf bv = frag_cat(vr + hi * 8, vr + 16 + hi * 8);
      o[t] = __builtin_amdgcn_wmma_f32_16x16x32_bf16(false, ap, false, bv,
                                                     (short)0, o[t], false, false);
    }
  }

  // ---- normalize + write f32 output ----
#pragma unroll
  for (int t = 0; t < 4; ++t)
#pragma unroll
    for (int r = 0; r < 8; ++r) {
      const int row = qb + wid * 16 + hi * 8 + r;
      const int col = h * DH + t * 16 + lo;
      O[((size_t)n * S + row) * D + col] = o[t][r] / l_i[r];
    }
}

// ---------------------------------------------------------------------------
// Host orchestration
// ---------------------------------------------------------------------------
extern "C" void kernel_launch(void* const* d_in, const int* in_sizes, int n_in,
                              void* d_out, int out_size, void* d_ws, size_t ws_size,
                              hipStream_t stream) {
  const int  Nb = 2, S = 2048, D = 512, H = 2048;
  const size_t MT = (size_t)Nb * S;               // 4096 rows

  const float* x = (const float*)d_in[0];
  const float* W1[3] = { (const float*)d_in[1], (const float*)d_in[5], (const float*)d_in[9]  };
  const float* b1[3] = { (const float*)d_in[2], (const float*)d_in[6], (const float*)d_in[10] };
  const float* W2[3] = { (const float*)d_in[3], (const float*)d_in[7], (const float*)d_in[11] };
  const float* b2[3] = { (const float*)d_in[4], (const float*)d_in[8], (const float*)d_in[12] };

  // workspace layout (bf16 buffers), ~44 MB total
  char* ws = (char*)d_ws;
  __bf16* xb = (__bf16*)ws;                 ws += MT * D * sizeof(__bf16);
  __bf16* w1t[3], *w2t[3], *qkv[3];
  for (int i = 0; i < 3; ++i) { w1t[i] = (__bf16*)ws; ws += (size_t)H * D * sizeof(__bf16); }
  for (int i = 0; i < 3; ++i) { w2t[i] = (__bf16*)ws; ws += (size_t)D * H * sizeof(__bf16); }
  for (int i = 0; i < 3; ++i) { qkv[i] = (__bf16*)ws; ws += MT * D * sizeof(__bf16); }
  __bf16* hid = (__bf16*)ws;                ws += MT * H * sizeof(__bf16);

  float* out_k = (float*)d_out;
  float* out_v = out_k + MT * D;
  float* out_o = out_v + MT * D;

  // 1) precision conversion (+ weight transpose to [N][K] for WMMA B tiles)
  {
    int n = (int)(MT * D);
    cvt_bf16_kernel<<<(n + 255) / 256, 256, 0, stream>>>(x, xb, n);
    int nw = D * H;
    for (int i = 0; i < 3; ++i) {
      cvt_bf16_t_kernel<<<(nw + 255) / 256, 256, 0, stream>>>(W1[i], w1t[i], D, H);
      cvt_bf16_t_kernel<<<(nw + 255) / 256, 256, 0, stream>>>(W2[i], w2t[i], H, D);
    }
  }

  // 2) QKV projection GEMMs (hid buffer reused sequentially; stream-ordered)
  dim3 blk(256);
  dim3 g1(H / BN, (unsigned)(MT / BM));   // 16 x 32
  dim3 g2(D / BN, (unsigned)(MT / BM));   //  4 x 32
  for (int i = 0; i < 3; ++i) {
    // hidden = gelu(x @ W1 + b1)         M=4096 N=2048 K=512
    gemm_bf16_kernel<<<g1, blk, 0, stream>>>(xb, w1t[i], b1[i], hid, nullptr,
                                             (int)MT, H, D, /*gelu=*/1);
    // proj = hidden @ W2 + b2            M=4096 N=512  K=2048
    float* f32out = (i == 1) ? out_k : (i == 2) ? out_v : nullptr;
    gemm_bf16_kernel<<<g2, blk, 0, stream>>>(hid, w2t[i], b2[i], qkv[i], f32out,
                                             (int)MT, D, H, /*gelu=*/(i == 2) ? 1 : 0);
  }

  // 3) causal flash attention: grid (q-tiles, heads, batch)
  dim3 ga(S / AQ, 8, Nb);
  attn_kernel<<<ga, 128, 0, stream>>>(qkv[0], qkv[1], qkv[2], out_o, S, D);
}
